// SubLSTM_29557964931761
// MI455X (gfx1250) — compile-verified, matrix operand order checked by reading, not run on hardware
//
#include <hip/hip_runtime.h>
#include <hip/hip_bf16.h>

// Problem constants (from reference)
#define G_  8
#define D_  20
#define H4_ 80      // 4*D
#define B_  64
#define T_  2000
#define F_  160     // G*D
#define BT_ (B_*T_) // 128000
#define N1_ 640     // 4*F
#define LSTM_THREADS 160   // 5 waves: one per 16-col gate tile
#define MLP_THREADS  256   // 8 waves
#define PRE_TT 8           // timesteps per block in the precompute kernel

typedef __attribute__((ext_vector_type(16))) __bf16 v16bf;
typedef __attribute__((ext_vector_type(8)))  __bf16 v8bf;
typedef __attribute__((ext_vector_type(8)))  float  v8f;
typedef __attribute__((ext_vector_type(4)))  unsigned int v4u;
typedef __attribute__((ext_vector_type(8)))  int v8i;
typedef __attribute__((ext_vector_type(4)))  int v4i;

// Tensor Data Mover availability (toolchain-dependent arity handled below)
#if defined(__has_builtin)
#  if __has_builtin(__builtin_amdgcn_tensor_load_to_lds) && \
      __has_builtin(__builtin_amdgcn_s_wait_tensorcnt)
#    define HAVE_TDM 1
#  endif
#endif
#ifndef HAVE_TDM
#  define HAVE_TDM 0
#endif

// 16-bit A/B-matrix 16x32 layout (ISA 7.12.2): VGPR j, lane-half h -> K base
__device__ __forceinline__ int wmma_k(int j, int half) {
    return (j < 4) ? (half * 8 + j * 2) : (16 + half * 8 + (j - 4) * 2);
}
// branch-free fast activations (v_exp_f32 + v_rcp_f32)
__device__ __forceinline__ float sigf(float x) {
    return __builtin_amdgcn_rcpf(1.f + __expf(-x));
}
__device__ __forceinline__ float tanh_fast(float x) {
    return 1.f - 2.f * __builtin_amdgcn_rcpf(1.f + __expf(2.f * x));
}
// build a 16-elem A/B fragment from a K-padded row base (two 16B chunks)
__device__ __forceinline__ v16bf frag_from_row(const __bf16* row, int half) {
    v8bf lo = *(const v8bf*)(row + half * 8);        // k = 8h..8h+7     (vgpr 0-3)
    v8bf hi = *(const v8bf*)(row + 16 + half * 8);   // k = 16+8h..23+8h (vgpr 4-7)
    v16bf a;
#pragma unroll
    for (int e = 0; e < 8; ++e) { a[e] = lo[e]; a[8 + e] = hi[e]; }
    return a;
}

// Fragment-layout gate buffer: [G][B/16][T][5 ntiles][32 lanes][8 f32]
#define XW_TILE_STRIDE 256                    // floats per (ntile) fragment
#define XW_T_STRIDE    (5 * XW_TILE_STRIDE)   // floats per timestep per block-row

#if HAVE_TDM
// Issue a 1-row TDM load: XW_T_STRIDE f32 (5120B contiguous) global -> LDS.
// D# per CDNA5 ISA. One call == one DMA (wave op).
__device__ __forceinline__ void tdm_load_row(const float* gsrc, unsigned int lds_addr) {
    unsigned long long ga = (unsigned long long)(const void*)gsrc;
    v4u g0;
    g0.x = 1u;                                         // count=1 (valid user D#)
    g0.y = lds_addr;                                   // lds_addr (bytes)
    g0.z = (unsigned int)ga;                           // global_addr[31:0]
    g0.w = (unsigned int)((ga >> 32) & 0x1FFFFFFu)     // global_addr[56:32]
         | (2u << 30);                                 // type=2 ("image")
    v8i g1;
    g1[0] = (int)(2u << 16);                           // data_size=2 -> 4 bytes
    g1[1] = (int)((XW_T_STRIDE & 0xFFFFu) << 16);      // tensor_dim0[15:0]
    g1[2] = (int)((XW_T_STRIDE >> 16) | (1u << 16));   // tensor_dim0[31:16] | tensor_dim1=1
    g1[3] = (int)((unsigned)XW_T_STRIDE << 16);        // tile_dim0 = 1280 elems
    g1[4] = 1;                                         // tile_dim1=1, tile_dim2=0
    g1[5] = XW_T_STRIDE;                               // tensor_dim0_stride
    g1[6] = 0;
    g1[7] = 0;
    v4i z4 = {0, 0, 0, 0};
#if __clang_major__ >= 23
    v8i z8 = {0, 0, 0, 0, 0, 0, 0, 0};
    __builtin_amdgcn_tensor_load_to_lds(g0, g1, z4, z4, z8, 0);
#else
    __builtin_amdgcn_tensor_load_to_lds(g0, g1, z4, z4, 0);
#endif
}
#endif

// ---------------------------------------------------------------------------
// Precompute xw = input @ Wih^T + (b_ih + b_hh), stored directly in WMMA
// C-fragment layout so the sequential kernel reads 32B/lane, coalesced.
// Fully parallel over (group, batch-tile, t). LAYER=0: input x f32 [B,T,G,D];
// LAYER=1: input hin bf16 [G,B,T,D].
// ---------------------------------------------------------------------------
template <int LAYER>
__global__ __launch_bounds__(LSTM_THREADS) void gates_pre_kernel(
    const float* __restrict__ x, const __bf16* __restrict__ hin,
    const float* __restrict__ Wih,
    const float* __restrict__ bih, const float* __restrict__ bhh,
    float* __restrict__ xw)
{
    __shared__ __bf16 x_s[16][32];   // K-padded input tile

    const int grp    = blockIdx.x;
    const int mtile  = blockIdx.y;
    const int tchunk = blockIdx.z;
    const int tid  = threadIdx.x;
    const int wave = tid >> 5;       // ntile 0..4
    const int lane = tid & 31;
    const int half = lane >> 4;
    const int lm   = lane & 15;

    for (int i = tid; i < 16 * 32; i += LSTM_THREADS)
        x_s[i >> 5][i & 31] = (__bf16)0.f;

    // Wih B-fragment + folded bias
    const int n_out = wave * 16 + lm;
    v16bf b_ih;
#pragma unroll
    for (int j = 0; j < 8; ++j) {
        int k0 = wmma_k(j, half);
#pragma unroll
        for (int e = 0; e < 2; ++e) {
            int k = k0 + e;
            b_ih[2 * j + e] = (__bf16)((k < D_) ? Wih[(grp * H4_ + n_out) * D_ + k] : 0.f);
        }
    }
    const float bias = bih[grp * H4_ + n_out] + bhh[grp * H4_ + n_out];

    for (int tt = 0; tt < PRE_TT; ++tt) {
        const int t = tchunk * PRE_TT + tt;
        // stage input tile (coalesced, only d<20 written; pad cols stay zero)
#pragma unroll 2
        for (int i = tid; i < 16 * D_; i += LSTM_THREADS) {
            int bl = i / D_, d = i % D_;
            float v;
            if (LAYER == 0)
                v = x[((size_t)(mtile * 16 + bl) * T_ + t) * (G_ * D_) + grp * D_ + d];
            else
                v = (float)hin[(((size_t)grp * B_ + mtile * 16 + bl) * T_ + t) * D_ + d];
            x_s[bl][d] = (__bf16)v;
        }
        __syncthreads();

        v16bf a = frag_from_row(&x_s[lm][0], half);
        v8f acc;
#pragma unroll
        for (int r = 0; r < 8; ++r) acc[r] = bias;
        acc = __builtin_amdgcn_wmma_f32_16x16x32_bf16(false, a, false, b_ih,
                                                      (short)0, acc, false, false);
        float* dst = xw + (((size_t)(grp * 4 + mtile) * T_ + t) * 5 + wave) * XW_TILE_STRIDE
                        + lane * 8;
        *(v8f*)dst = acc;
        __syncthreads();   // before next stage overwrites x_s
    }
}

// ---------------------------------------------------------------------------
// Sequential LSTM recurrence. One block per (group, batch-tile of 16).
// Per step: consume precomputed gate fragment (TDM double-buffered in LDS,
// or global load + prefetch fallback), ONE WMMA (h @ Whh^T, C = fragment),
// gate scatter, cell update.
// LAYER=0: writes h -> hout [G,B,T,D] bf16 (feeds layer-1 precompute)
// LAYER=1: writes h -> fbh  [B,T,F] bf16  (residual added later in MLP)
// ---------------------------------------------------------------------------
template <int LAYER>
__global__ __launch_bounds__(LSTM_THREADS) void lstm_seq_kernel(
    const float* __restrict__ xw,
    const float* __restrict__ Whh,
    __bf16* __restrict__ hout)
{
    __shared__ __bf16 h_s[16][32];   // K-padded h state
    __shared__ float  c_s[16][D_];
    __shared__ float  g_s[16][H4_];
#if HAVE_TDM
    __shared__ float  xw_s[2][XW_T_STRIDE];   // double-buffered gate tile (2x5KB)
#endif

    const int grp   = blockIdx.x;
    const int mtile = blockIdx.y;
    const int tid   = threadIdx.x;
    const int wave  = tid >> 5;
    const int lane  = tid & 31;
    const int half  = lane >> 4;
    const int lm    = lane & 15;

    for (int i = tid; i < 16 * 32; i += LSTM_THREADS) h_s[i >> 5][i & 31] = (__bf16)0.f;
    for (int i = tid; i < 16 * D_; i += LSTM_THREADS) c_s[i / D_][i % D_] = 0.f;

    // Whh B-fragment (constant over time)
    const int n_out = wave * 16 + lm;
    v16bf b_hh;
#pragma unroll
    for (int j = 0; j < 8; ++j) {
        int k0 = wmma_k(j, half);
#pragma unroll
        for (int e = 0; e < 2; ++e) {
            int k = k0 + e;
            b_hh[2 * j + e] = (__bf16)((k < D_) ? Whh[(grp * H4_ + n_out) * D_ + k] : 0.f);
        }
    }

    const float* xw_blk = xw + (size_t)(grp * 4 + mtile) * T_ * XW_T_STRIDE;
#if HAVE_TDM
    if (wave == 0)
        tdm_load_row(xw_blk, (unsigned int)(size_t)&xw_s[0][0]);   // tile for t=0
#endif
    __syncthreads();

    for (int t = 0; t < T_; ++t) {
#if HAVE_TDM
        const int buf = t & 1;
        if (wave == 0) __builtin_amdgcn_s_wait_tensorcnt(0);
        __syncthreads();                       // xw_s[buf] complete & visible
        if (wave == 0 && t + 1 < T_)
            tdm_load_row(xw_blk + (size_t)(t + 1) * XW_T_STRIDE,
                         (unsigned int)(size_t)&xw_s[buf ^ 1][0]);
        v8f cfrag = *(const v8f*)&xw_s[buf][(wave * 32 + lane) * 8];
#else
        const float* xwp = xw_blk + (size_t)t * XW_T_STRIDE
                         + (size_t)wave * XW_TILE_STRIDE + lane * 8;
        v8f cfrag = *(const v8f*)xwp;                       // 32B/lane, coalesced
        if (t + 1 < T_)
            __builtin_prefetch(xwp + XW_T_STRIDE, 0, 3);    // global_prefetch_b8
#endif

        v16bf a_h = frag_from_row(&h_s[lm][0], half);
        v8f acc = __builtin_amdgcn_wmma_f32_16x16x32_bf16(false, a_h, false, b_hh,
                                                          (short)0, cfrag, false, false);
        // C layout: VGPR r -> row (half*8 + r), col lm
#pragma unroll
        for (int r = 0; r < 8; ++r)
            g_s[half * 8 + r][n_out] = acc[r];
        __syncthreads();

        // cell update: 320 cells over 160 threads
#pragma unroll 2
        for (int c2 = tid; c2 < 16 * D_; c2 += LSTM_THREADS) {
            int bl = c2 / D_, d = c2 % D_;
            float ig = sigf(g_s[bl][d]);
            float fg = sigf(g_s[bl][D_ + d]);
            float gg = tanh_fast(g_s[bl][2 * D_ + d]);
            float og = sigf(g_s[bl][3 * D_ + d]);
            float c  = fg * c_s[bl][d] + ig * gg;
            float h  = og * tanh_fast(c);
            c_s[bl][d] = c;
            h_s[bl][d] = (__bf16)h;
            int bg = mtile * 16 + bl;
            if (LAYER == 0)
                hout[(((size_t)grp * B_ + bg) * T_ + t) * D_ + d] = (__bf16)h;
            else
                hout[((size_t)bg * T_ + t) * F_ + grp * D_ + d] = (__bf16)h;
        }
        __syncthreads();
    }
}

// ---------------------------------------------------------------------------
// Pack f32 W[N,K] into bf16 WMMA B-fragments: [nt][kt][lane][16], 32B/lane.
// ---------------------------------------------------------------------------
__global__ __launch_bounds__(MLP_THREADS) void pack_w_kernel(
    const float* __restrict__ W, __bf16* __restrict__ Wp, int N, int K)
{
    int idx = blockIdx.x * blockDim.x + threadIdx.x;
    int numK = K >> 5;
    int total = (N >> 4) * numK * 32;
    if (idx >= total) return;
    int lane = idx & 31;
    int fk = (idx >> 5) % numK;
    int fn = (idx >> 5) / numK;
    int half = lane >> 4, ln = lane & 15;
    int n = fn * 16 + ln;
    v16bf out;
#pragma unroll
    for (int j = 0; j < 8; ++j) {
        int k = fk * 32 + wmma_k(j, half);
        out[2 * j]     = (__bf16)W[n * K + k];
        out[2 * j + 1] = (__bf16)W[n * K + k + 1];
    }
    *(v16bf*)(Wp + (size_t)idx * 16) = out;
}

// ---------------------------------------------------------------------------
// Fused MLP, rows [32*tm, 32*tm+32):
//   stage fb = lstm_h + x (both rows contiguous 160-wide) -> LDS bf16
//   GEMM1: h1 = relu(fb @ W1^T + b1) -> LDS slab 32x640 bf16 (40KB)
//   GEMM2: out = h1 @ W2^T + b2      -> f32 global
// ---------------------------------------------------------------------------
__global__ __launch_bounds__(MLP_THREADS) void mlp_fused_kernel(
    const __bf16* __restrict__ fbh, const float* __restrict__ x,
    const __bf16* __restrict__ W1p, const float* __restrict__ b1,
    const __bf16* __restrict__ W2p, const float* __restrict__ b2,
    float* __restrict__ out)
{
    __shared__ __bf16 fb_s[32][F_];    // 10 KB
    __shared__ __bf16 h1_s[32][N1_];   // 40 KB

    const int tid  = threadIdx.x;
    const int wave = tid >> 5;
    const int lane = tid & 31;
    const int half = lane >> 4;
    const int lm   = lane & 15;
    const int tm   = blockIdx.x;

    // stage residual slab: fb = h + x (coalesced; both are row-contiguous)
    for (int i = tid; i < 32 * F_; i += MLP_THREADS) {
        int r = i / F_, f = i % F_;
        size_t row = (size_t)tm * 32 + r;
        float v = (float)fbh[row * F_ + f] + x[row * F_ + f];
        fb_s[r][f] = (__bf16)v;
    }
    __syncthreads();

    // ---- GEMM1: [32,160] @ W1^T -> relu -> h1_s [32,640] ----
    {
        const int numK = F_ >> 5;      // 5
        for (int tn = wave; tn < (N1_ >> 5); tn += 8) {
            v8f acc[2][2];
#pragma unroll
            for (int i = 0; i < 2; ++i)
#pragma unroll
                for (int j = 0; j < 2; ++j)
#pragma unroll
                    for (int r = 0; r < 8; ++r) acc[i][j][r] = 0.f;
#pragma unroll
            for (int kt = 0; kt < numK; ++kt) {
                v16bf a[2], b[2];
#pragma unroll
                for (int i = 0; i < 2; ++i)
                    a[i] = frag_from_row(&fb_s[i * 16 + lm][kt * 32], half);
#pragma unroll
                for (int j = 0; j < 2; ++j) {
                    size_t fidx = ((size_t)(tn * 2 + j) * numK + kt) * 32 + lane;
                    b[j] = *(const v16bf*)(W1p + fidx * 16);
                }
#pragma unroll
                for (int i = 0; i < 2; ++i)
#pragma unroll
                    for (int j = 0; j < 2; ++j)
                        acc[i][j] = __builtin_amdgcn_wmma_f32_16x16x32_bf16(
                            false, a[i], false, b[j], (short)0, acc[i][j], false, false);
            }
#pragma unroll
            for (int i = 0; i < 2; ++i)
#pragma unroll
                for (int j = 0; j < 2; ++j) {
                    int n = tn * 32 + j * 16 + lm;
                    float bv = b1[n];
#pragma unroll
                    for (int r = 0; r < 8; ++r) {
                        float v = acc[i][j][r] + bv;
                        v = v > 0.f ? v : 0.f;
                        h1_s[i * 16 + half * 8 + r][n] = (__bf16)v;
                    }
                }
        }
    }
    __syncthreads();

    // ---- GEMM2: [32,640] @ W2^T -> [32,160] f32 out ----
    {
        const int numK = N1_ >> 5;     // 20
        for (int tn = wave; tn < (F_ >> 5); tn += 8) {    // 5 tiles, waves 0..4
            v8f acc[2][2];
#pragma unroll
            for (int i = 0; i < 2; ++i)
#pragma unroll
                for (int j = 0; j < 2; ++j)
#pragma unroll
                    for (int r = 0; r < 8; ++r) acc[i][j][r] = 0.f;
            for (int kt = 0; kt < numK; ++kt) {
                v16bf a[2], b[2];
#pragma unroll
                for (int i = 0; i < 2; ++i)
                    a[i] = frag_from_row(&h1_s[i * 16 + lm][kt * 32], half);
#pragma unroll
                for (int j = 0; j < 2; ++j) {
                    size_t fidx = ((size_t)(tn * 2 + j) * numK + kt) * 32 + lane;
                    b[j] = *(const v16bf*)(W2p + fidx * 16);
                }
#pragma unroll
                for (int i = 0; i < 2; ++i)
#pragma unroll
                    for (int j = 0; j < 2; ++j)
                        acc[i][j] = __builtin_amdgcn_wmma_f32_16x16x32_bf16(
                            false, a[i], false, b[j], (short)0, acc[i][j], false, false);
            }
#pragma unroll
            for (int i = 0; i < 2; ++i)
#pragma unroll
                for (int j = 0; j < 2; ++j) {
                    int n = tn * 32 + j * 16 + lm;
                    float bv = b2[n];
#pragma unroll
                    for (int r = 0; r < 8; ++r) {
                        int m = tm * 32 + i * 16 + half * 8 + r;
                        out[(size_t)m * F_ + n] = acc[i][j][r] + bv;
                    }
                }
        }
    }
}

// ---------------------------------------------------------------------------
extern "C" void kernel_launch(void* const* d_in, const int* in_sizes, int n_in,
                              void* d_out, int out_size, void* d_ws, size_t ws_size,
                              hipStream_t stream) {
    const float* x     = (const float*)d_in[0];
    const float* Wih0  = (const float*)d_in[1];
    const float* Whh0  = (const float*)d_in[2];
    const float* bih0  = (const float*)d_in[3];
    const float* bhh0  = (const float*)d_in[4];
    const float* Wih1  = (const float*)d_in[5];
    const float* Whh1  = (const float*)d_in[6];
    const float* bih1  = (const float*)d_in[7];
    const float* bhh1  = (const float*)d_in[8];
    const float* W1    = (const float*)d_in[9];
    const float* b1    = (const float*)d_in[10];
    const float* W2    = (const float*)d_in[11];
    const float* b2    = (const float*)d_in[12];
    float* out = (float*)d_out;

    // workspace carve-out (256B aligned slices)
    size_t off = 0;
    auto carve = [&](size_t bytes) {
        void* p = (char*)d_ws + off;
        off += (bytes + 255) & ~(size_t)255;
        return p;
    };
    float*  xw  = (float*)carve((size_t)G_ * 4 * T_ * 5 * XW_TILE_STRIDE * sizeof(float));
    __bf16* h0  = (__bf16*)carve((size_t)G_ * B_ * T_ * D_ * sizeof(__bf16));
    __bf16* fbh = (__bf16*)carve((size_t)BT_ * F_ * sizeof(__bf16));
    __bf16* W1p = (__bf16*)carve((size_t)(N1_ / 16) * (F_ / 32) * 32 * 16 * sizeof(__bf16));
    __bf16* W2p = (__bf16*)carve((size_t)(F_ / 16) * (N1_ / 32) * 32 * 16 * sizeof(__bf16));

    // 1) pack MLP weights into WMMA fragment layout
    {
        int tot1 = (N1_ / 16) * (F_ / 32) * 32;
        int tot2 = (F_ / 16) * (N1_ / 32) * 32;
        pack_w_kernel<<<(tot1 + MLP_THREADS - 1) / MLP_THREADS, MLP_THREADS, 0, stream>>>(
            W1, W1p, N1_, F_);
        pack_w_kernel<<<(tot2 + MLP_THREADS - 1) / MLP_THREADS, MLP_THREADS, 0, stream>>>(
            W2, W2p, F_, N1_);
    }

    // 2) layer 0: parallel input-GEMM precompute, then sequential recurrence
    {
        dim3 pre_grid(G_, B_ / 16, T_ / PRE_TT);
        dim3 seq_grid(G_, B_ / 16);
        gates_pre_kernel<0><<<pre_grid, LSTM_THREADS, 0, stream>>>(
            x, nullptr, Wih0, bih0, bhh0, xw);
        lstm_seq_kernel<0><<<seq_grid, LSTM_THREADS, 0, stream>>>(xw, Whh0, h0);
        // 3) layer 1 (input = h0), xw buffer reused
        gates_pre_kernel<1><<<pre_grid, LSTM_THREADS, 0, stream>>>(
            nullptr, h0, Wih1, bih1, bhh1, xw);
        lstm_seq_kernel<1><<<seq_grid, LSTM_THREADS, 0, stream>>>(xw, Whh1, fbh);
    }

    // 4) fused MLP (residual add folded into staging), 4000 row-slab blocks
    mlp_fused_kernel<<<BT_ / 32, MLP_THREADS, 0, stream>>>(
        fbh, x, W1p, b1, W2p, b2, out);

    (void)in_sizes; (void)n_in; (void)out_size; (void)ws_size;
}
